// MatchingAttention_36679020708321
// MI455X (gfx1250) — matrix operand order repaired and checked by reading, not compiled, not run on hardware
//
#include <hip/hip_runtime.h>
#include <hip/hip_bf16.h>

// ---------------------------------------------------------------------------
// MatchingAttention for MI455X (gfx1250, wave32, WMMA bf16)
// Shapes: S=64, L=64, B=16, D_IN=1024, D_ATTN=512
// ---------------------------------------------------------------------------

#define S_DIM   64
#define L_DIM   64
#define B_DIM   16
#define D_IN    1024
#define D_ATTN  512
#define LB      (L_DIM * B_DIM)            // 1024
#define ROWS_C  (S_DIM * LB)               // 65536
#define ROWS_Q  LB                         // 1024

typedef __attribute__((ext_vector_type(16))) __bf16 v16bf;
typedef __attribute__((ext_vector_type(2)))  __bf16 v2bf;
typedef __attribute__((ext_vector_type(8)))  float  v8f;

struct U8x32 { uint4 a, b; };   // 32 bytes, reinterpreted as v16bf

// fp32 pair -> packed bf16 (RNE). Prefer hardware v_cvt_pk_bf16_f32;
// otherwise rely on native fptrunc float->bfloat lowering.
static __device__ __forceinline__ unsigned pk2bf(float lo, float hi) {
#if __has_builtin(__builtin_amdgcn_cvt_pk_bf16_f32)
    v2bf r = __builtin_amdgcn_cvt_pk_bf16_f32(lo, hi);
    return __builtin_bit_cast(unsigned, r);
#else
    v2bf r;
    r[0] = (__bf16)lo;
    r[1] = (__bf16)hi;
    return __builtin_bit_cast(unsigned, r);
#endif
}

// ---------------------------------------------------------------------------
// Kernel 0: convert Wc / Wq (fp32 [512][1024]) to packed bf16 in workspace
// ---------------------------------------------------------------------------
__global__ void cvt_kernel(const float* __restrict__ Wc,
                           const float* __restrict__ Wq,
                           unsigned* __restrict__ Wcb,
                           unsigned* __restrict__ Wqb) {
    int i = blockIdx.x * blockDim.x + threadIdx.x;   // 262144 u32 per matrix
    if (i < (D_ATTN * D_IN) / 2) {
        const float2 c = reinterpret_cast<const float2*>(Wc)[i];
        const float2 q = reinterpret_cast<const float2*>(Wq)[i];
        Wcb[i] = pk2bf(c.x, c.y);
        Wqb[i] = pk2bf(q.x, q.y);
    }
}

// ---------------------------------------------------------------------------
// WMMA GEMM:  rows x 512 = (rows x 1024) * W^T,  W bf16 [512][1024] row-major
//   SCORE=false : out[row*512+n] = dot + bias[n]                 (query proj)
//   SCORE=true  : score[row]     = sum_n v[n]*tanh(dot+bias[n]+qp[row%1024][n])
// Block: 256 threads = 8 waves = 2 (M subtiles of 16) x 4 (N groups of 128).
// M tile = 32 rows; K staged through LDS in 64-wide bf16 chunks (16 iters).
// Two explicit K-fragment passes; B double-buffered (parity registers).
// ---------------------------------------------------------------------------
template <bool SCORE>
__global__ void
__attribute__((amdgpu_waves_per_eu(1)))
gemm_kernel(const float* __restrict__ X,           // rows x 1024
            const unsigned short* __restrict__ Wb, // bf16 512x1024
            const float* __restrict__ bias,        // [512]
            const float* __restrict__ qp,          // [1024][512]
            const float* __restrict__ vvec,        // [512]
            float* __restrict__ out) {
    __shared__ unsigned short Ald[32 * 64];   // 32 rows x 64 k, bf16 (4 KB)
    __shared__ float scAcc[32];

    const int tid    = threadIdx.x;
    const int wave   = tid >> 5;
    const int lane   = tid & 31;
    const int lane16 = lane & 15;
    const int laneHi = lane >> 4;
    const int mi     = wave >> 2;       // 0..1  : M subtile
    const int ng     = wave & 3;        // 0..3  : N group (128 cols)
    const int rowBase = blockIdx.x * 32;

    if (SCORE && tid < 32) scAcc[tid] = 0.0f;

    v8f acc[8];
#pragma unroll
    for (int t = 0; t < 8; ++t) acc[t] = (v8f){0,0,0,0,0,0,0,0};

    const int stRow = tid >> 3;          // 0..31 : staging row
    const int stK   = (tid & 7) * 8;     // 0..56 : staging k octet
    const float* __restrict__ Xst = X + (size_t)(rowBase + stRow) * D_IN + stK;

    // prefetch first A chunk
    float4 f0 = *reinterpret_cast<const float4*>(Xst);
    float4 f1 = *reinterpret_cast<const float4*>(Xst + 4);

    for (int kb = 0; kb < D_IN; kb += 64) {
        __syncthreads();
        // ---- stage A tile (32 rows x 64 k) as bf16 into LDS ----
        {
            uint4 p;
            p.x = pk2bf(f0.x, f0.y);
            p.y = pk2bf(f0.z, f0.w);
            p.z = pk2bf(f1.x, f1.y);
            p.w = pk2bf(f1.z, f1.w);
            *reinterpret_cast<uint4*>(&Ald[stRow * 64 + stK]) = p;
        }
        __syncthreads();

        // ---- prefetch next A chunk (latency hidden behind 16 WMMAs) ----
        if (kb + 64 < D_IN) {
            f0 = *reinterpret_cast<const float4*>(Xst + kb + 64);
            f1 = *reinterpret_cast<const float4*>(Xst + kb + 68);
        }

        // ---- A fragments (16x32 bf16: lane halves K {0-7,16-23}/{8-15,24-31}) ----
        const int am = mi * 16 + lane16;
        const uint4* Ap = reinterpret_cast<const uint4*>(&Ald[am * 64 + laneHi * 8]);
        U8x32 a0, a1;
        a0.a = Ap[0];     // k = laneHi*8 + 0..7
        a0.b = Ap[2];     // k = laneHi*8 + 16..23
        a1.a = Ap[4];     // k = 32 + laneHi*8 + 0..7
        a1.b = Ap[6];     // k = 32 + laneHi*8 + 16..23
        const v16bf av0 = __builtin_bit_cast(v16bf, a0);
        const v16bf av1 = __builtin_bit_cast(v16bf, a1);

        // ---- two K-fragment passes over the 8 N tiles, each pipelined ----
        const unsigned short* __restrict__ Bbase =
            Wb + (size_t)(ng * 128 + lane16) * D_IN + kb + laneHi * 8;
#pragma unroll
        for (int kk = 0; kk < 2; ++kk) {
            const v16bf av = kk ? av1 : av0;
            const unsigned short* __restrict__ Bk = Bbase + kk * 32;
            U8x32 buf[2];
            {
                const uint4* Bp = reinterpret_cast<const uint4*>(Bk);
                buf[0].a = Bp[0]; buf[0].b = Bp[2];
            }
#pragma unroll
            for (int t = 0; t < 8; ++t) {
                if (t < 7) {
                    const uint4* Bp = reinterpret_cast<const uint4*>(
                        Bk + (size_t)(t + 1) * 16 * D_IN);
                    buf[(t + 1) & 1].a = Bp[0];
                    buf[(t + 1) & 1].b = Bp[2];
                }
                acc[t] = __builtin_amdgcn_wmma_f32_16x16x32_bf16(
                    false, av, false, __builtin_bit_cast(v16bf, buf[t & 1]),
                    (short)0, acc[t], false, false);
            }
        }
    }

    // ---- epilogue: C layout lanes 0-15 -> M=r, lanes 16-31 -> M=r+8 ----
    if constexpr (!SCORE) {
#pragma unroll
        for (int t = 0; t < 8; ++t) {
            const int n = ng * 128 + t * 16 + lane16;
            const float bn = bias[n];
#pragma unroll
            for (int r = 0; r < 8; ++r) {
                const int m = mi * 16 + r + laneHi * 8;
                out[(size_t)(rowBase + m) * D_ATTN + n] = acc[t][r] + bn;
            }
        }
    } else {
        float part[8];
#pragma unroll
        for (int r = 0; r < 8; ++r) part[r] = 0.0f;
#pragma unroll
        for (int t = 0; t < 8; ++t) {
            const int n  = ng * 128 + t * 16 + lane16;
            const float bn = bias[n];
            const float vn = vvec[n];
#pragma unroll
            for (int r = 0; r < 8; ++r) {
                const int m    = mi * 16 + r + laneHi * 8;
                const int qrow = (rowBase + m) & (LB - 1);
                const float x  = acc[t][r] + bn + qp[(size_t)qrow * D_ATTN + n];
                part[r] += vn * tanhf(x);
            }
        }
        // reduce across the 16 lanes of each half (same row set per half)
#pragma unroll
        for (int r = 0; r < 8; ++r) {
#pragma unroll
            for (int msk = 8; msk >= 1; msk >>= 1)
                part[r] += __shfl_xor(part[r], msk, 32);
        }
        if (lane16 == 0) {
#pragma unroll
            for (int r = 0; r < 8; ++r)
                atomicAdd(&scAcc[mi * 16 + r + laneHi * 8], part[r]);
        }
        __syncthreads();
        if (tid < 32) out[rowBase + tid] = scAcc[tid];
    }
}

// ---------------------------------------------------------------------------
// Softmax over S (=64) per (l,b) column; applies mask; writes alpha.
// scores/mask flat index: s*1024 + lb
// ---------------------------------------------------------------------------
__global__ void softmax_kernel(const float* __restrict__ sc,
                               const int* __restrict__ mask,
                               float* __restrict__ alpha) {
    const int lb = blockIdx.x * blockDim.x + threadIdx.x;
    if (lb >= LB) return;
    float vals[S_DIM];
    float mx = -3.4e38f;
#pragma unroll
    for (int s = 0; s < S_DIM; ++s) {
        float x = sc[s * LB + lb];
        if (mask[s * LB + lb] == 0) x = -1e9f;
        vals[s] = x;
        mx = fmaxf(mx, x);
    }
    float sum = 0.0f;
#pragma unroll
    for (int s = 0; s < S_DIM; ++s) {
        vals[s] = __expf(vals[s] - mx);
        sum += vals[s];
    }
    const float inv = 1.0f / sum;
#pragma unroll
    for (int s = 0; s < S_DIM; ++s)
        alpha[s * LB + lb] = vals[s] * inv;
}

// ---------------------------------------------------------------------------
// attn_output[lb][d] = sum_s alpha[s][lb] * context[s*1024+lb][d]
// one block per lb; float4 coalesced reads, stride 4 MB over s.
// ---------------------------------------------------------------------------
__global__ void wsum_kernel(const float* __restrict__ ctx,
                            const float* __restrict__ alpha,
                            float* __restrict__ out) {
    const int lb = blockIdx.x;
    const int t  = threadIdx.x;            // 256 threads, 4 floats each
    __shared__ float al[S_DIM];
    if (t < S_DIM) al[t] = alpha[t * LB + lb];
    __syncthreads();
    float4 acc = {0.0f, 0.0f, 0.0f, 0.0f};
    const float4* base = reinterpret_cast<const float4*>(
        ctx + (size_t)lb * D_IN + t * 4);
    const size_t sStride = (size_t)LB * D_IN / 4;   // float4 stride per s
#pragma unroll 4
    for (int s = 0; s < S_DIM; ++s) {
        const float4 c = base[s * sStride];
        const float a = al[s];
        acc.x += a * c.x; acc.y += a * c.y;
        acc.z += a * c.z; acc.w += a * c.w;
    }
    reinterpret_cast<float4*>(out + (size_t)lb * D_IN)[t] = acc;
}

// ---------------------------------------------------------------------------
extern "C" void kernel_launch(void* const* d_in, const int* in_sizes, int n_in,
                              void* d_out, int out_size, void* d_ws, size_t ws_size,
                              hipStream_t stream) {
    const float* context = (const float*)d_in[0];   // [64,64,16,1024]
    const float* query   = (const float*)d_in[1];   // [64,16,1024]
    const int*   mask    = (const int*)  d_in[2];   // [64,64,16]
    const float* Wc      = (const float*)d_in[3];   // [512,1024]
    const float* bc      = (const float*)d_in[4];   // [512]
    const float* Wq      = (const float*)d_in[5];   // [512,1024]
    const float* bq      = (const float*)d_in[6];   // [512]
    const float* vvec    = (const float*)d_in[7];   // [512]

    float* out_attn  = (float*)d_out;                      // 1048576 floats
    float* out_alpha = (float*)d_out + (size_t)LB * D_IN;  // 65536 floats

    // workspace layout
    unsigned short* Wcb = (unsigned short*)d_ws;                  // 1 MB bf16
    unsigned short* Wqb = Wcb + (size_t)D_ATTN * D_IN;            // 1 MB bf16
    float* qp     = (float*)(Wqb + (size_t)D_ATTN * D_IN);        // 2 MB
    float* scores = qp + (size_t)ROWS_Q * D_ATTN;                 // 256 KB

    // 0) weights -> bf16
    cvt_kernel<<<(D_ATTN * D_IN / 2 + 255) / 256, 256, 0, stream>>>(Wc, Wq,
        (unsigned*)Wcb, (unsigned*)Wqb);

    // 1) query projection: qp[lb][n] = query·Wq^T + bq
    gemm_kernel<false><<<ROWS_Q / 32, 256, 0, stream>>>(
        query, Wqb, bq, nullptr, nullptr, qp);

    // 2) fused context projection + tanh·v score reduction
    gemm_kernel<true><<<ROWS_C / 32, 256, 0, stream>>>(
        context, Wcb, bc, qp, vvec, scores);

    // 3) masked softmax over S -> alpha (second tuple output)
    softmax_kernel<<<LB / 256, 256, 0, stream>>>(scores, mask, out_alpha);

    // 4) alpha-weighted sum of context -> attn_output (first tuple output)
    wsum_kernel<<<LB, 256, 0, stream>>>(context, out_alpha, out_attn);
}
